// IrradianceModel_58875411693814
// MI455X (gfx1250) — compile-verified, matrix operand order checked by reading, not run on hardware
//
#include <hip/hip_runtime.h>

// ---------------------------------------------------------------------------
// MI455X (gfx1250) implementation of the DiT-style forward pass.
// All large matmuls -> v_wmma_f32_16x16x32_f16 with LDS-staged 64x64 tiles,
// fused bias / GELU / adaLN-gated-residual epilogues. Wave32 everywhere.
// ---------------------------------------------------------------------------

typedef _Float16 half_t;
typedef __attribute__((ext_vector_type(16))) _Float16 v16h;
typedef __attribute__((ext_vector_type(8)))  float    v8f;

constexpr int kD   = 512;
constexpr int kC   = 128;
constexpr int kL   = 12;
constexpr int kHX  = 4;
constexpr int kHP  = 8;
constexpr int kDH  = 8;
constexpr int kG   = 32;
constexpr int kTIN = 12;
constexpr int kOUT = 24;
constexpr int kB   = 4;
constexpr int kN   = 4096;
constexpr int kNS  = 128;
constexpr int kBN  = kB * kN;      // 16384 tokens
constexpr int kINRAW = 42;         // 2 + 24 + 16

// -------------------------------- helpers ----------------------------------

__device__ __forceinline__ float block_reduce_sum(float v, float* red, int bs) {
  int tid = threadIdx.x;
  red[tid] = v;
  __syncthreads();
  for (int s = bs >> 1; s > 0; s >>= 1) {
    if (tid < s) red[tid] += red[tid + s];
    __syncthreads();
  }
  float r = red[0];
  __syncthreads();
  return r;
}

__device__ __forceinline__ float gelu_exact(float x) {
  return 0.5f * x * (1.0f + erff(x * 0.70710678118654752f));
}

__device__ __forceinline__ float silu_f(float x) {
  return x / (1.0f + expf(-x));
}

// ------------------------------ WMMA GEMM ----------------------------------
// C[M,N] = A[M,K](f16 row-major) * Bt[N,K](f16, weights pre-transposed)
// Block: 128 threads = 4 waves; tile 64(M) x 64(N); K-step 32.
// Each wave owns a 32x32 subtile = 2x2 fragments of 16x16.
// EPI 0: Cf[m*N+n]     = acc + bias
// EPI 1: Ch[m*N+n]     = (f16) gelu(acc + bias)
// EPI 2: resid[m*N+n] += gate[(m>>12)*gate_ld + gate_ofs + n] * (acc + bias)

template <int EPI>
__global__ __launch_bounds__(128) void gemm_wmma(
    const half_t* __restrict__ A, const half_t* __restrict__ Bt,
    const float* __restrict__ bias,
    float* __restrict__ Cf, half_t* __restrict__ Ch,
    float* __restrict__ resid, const float* __restrict__ gate,
    int gate_ld, int gate_ofs, int M, int N, int K) {
  __shared__ __align__(16) half_t As[64][40];
  __shared__ __align__(16) half_t Bs[64][40];

  const int tid  = threadIdx.x;
  const int lane = tid & 31;
  const int wave = tid >> 5;
  const int wm   = (wave >> 1) * 32;
  const int wn   = (wave & 1) * 32;
  const int bm   = blockIdx.x * 64;
  const int bn   = blockIdx.y * 64;

  const int lr = tid >> 2;          // 0..31 (two rows per thread: lr, lr+32)
  const int lc = (tid & 3) * 8;     // 0,8,16,24 halves

  v8f acc[2][2] = {};

  // A-fragment lane mapping (ISA 7.12.2, 16-bit A 16x32):
  const int frow = lane & 15;
  const int kb   = (lane >> 4) * 8;
  int kofs[8];
#pragma unroll
  for (int v = 0; v < 8; ++v) kofs[v] = kb + ((v < 4) ? 2 * v : 16 + 2 * (v - 4));

  const half_t* Ap0 = A  + (size_t)(bm + lr)      * K + lc;
  const half_t* Ap1 = A  + (size_t)(bm + lr + 32) * K + lc;
  const half_t* Bp0 = Bt + (size_t)(bn + lr)      * K + lc;
  const half_t* Bp1 = Bt + (size_t)(bn + lr + 32) * K + lc;

  for (int k0 = 0; k0 < K; k0 += 32) {
    *reinterpret_cast<uint4*>(&As[lr][lc])      = *reinterpret_cast<const uint4*>(Ap0 + k0);
    *reinterpret_cast<uint4*>(&As[lr + 32][lc]) = *reinterpret_cast<const uint4*>(Ap1 + k0);
    *reinterpret_cast<uint4*>(&Bs[lr][lc])      = *reinterpret_cast<const uint4*>(Bp0 + k0);
    *reinterpret_cast<uint4*>(&Bs[lr + 32][lc]) = *reinterpret_cast<const uint4*>(Bp1 + k0);
    if (k0 + 32 < K) {
      __builtin_prefetch(Ap0 + k0 + 32, 0, 1);  // global_prefetch_b8
      __builtin_prefetch(Bp0 + k0 + 32, 0, 1);
    }
    __syncthreads();

    v16h afrag[2], bfrag[2];
#pragma unroll
    for (int i = 0; i < 2; ++i) {
#pragma unroll
      for (int v = 0; v < 8; ++v) {
        const int kp = kofs[v];
        afrag[i][2 * v]     = As[wm + 16 * i + frow][kp];
        afrag[i][2 * v + 1] = As[wm + 16 * i + frow][kp + 1];
        bfrag[i][2 * v]     = Bs[wn + 16 * i + frow][kp];
        bfrag[i][2 * v + 1] = Bs[wn + 16 * i + frow][kp + 1];
      }
    }
#pragma unroll
    for (int i = 0; i < 2; ++i)
#pragma unroll
      for (int j = 0; j < 2; ++j)
        acc[i][j] = __builtin_amdgcn_wmma_f32_16x16x32_f16(
            false, afrag[i], false, bfrag[j], (short)0, acc[i][j], false, false);
    __syncthreads();
  }

  // Epilogue. C/D layout: lane -> N = lane&15, VGPR r -> M = r + 8*(lane>=16).
  const int cn = lane & 15;
  const int rb = (lane >> 4) * 8;
#pragma unroll
  for (int i = 0; i < 2; ++i) {
#pragma unroll
    for (int j = 0; j < 2; ++j) {
#pragma unroll
      for (int r = 0; r < 8; ++r) {
        const int m = bm + wm + 16 * i + rb + r;
        const int n = bn + wn + 16 * j + cn;
        float v = acc[i][j][r];
        if (bias) v += bias[n];
        if (EPI == 0) {
          Cf[(size_t)m * N + n] = v;
        } else if (EPI == 1) {
          Ch[(size_t)m * N + n] = (half_t)gelu_exact(v);
        } else {
          const int b = m >> 12;  // 4096 tokens per batch
          resid[(size_t)m * N + n] += gate[b * gate_ld + gate_ofs + n] * v;
        }
      }
    }
  }
}

// -------------------------- packing kernels --------------------------------

// W[K,Nsrc] f32 -> Wt[Npad][Kpad] f16 (transposed, zero padded)
__global__ void pack_wt(const float* __restrict__ W, half_t* __restrict__ Wt,
                        int K, int Ns, int Kp, int Np) {
  int id = blockIdx.x * blockDim.x + threadIdx.x;
  if (id >= Np * Kp) return;
  int n = id / Kp, k = id % Kp;
  float v = (n < Ns && k < K) ? W[(size_t)k * Ns + n] : 0.0f;
  Wt[id] = (half_t)v;
}

// A[M,Ks] f32 -> A16[M,Kp] f16 (zero padded in K)
__global__ void pack_a(const float* __restrict__ in, half_t* __restrict__ out,
                       int M, int Ks, int Kp) {
  int id = blockIdx.x * blockDim.x + threadIdx.x;
  if (id >= M * Kp) return;
  int m = id / Kp, k = id % Kp;
  out[id] = (half_t)((k < Ks) ? in[(size_t)m * Ks + k] : 0.0f);
}

__global__ void pad_bias(const float* __restrict__ b, float* __restrict__ bp,
                         int Ns, int Np) {
  int i = threadIdx.x;
  if (i < Np) bp[i] = (i < Ns) ? b[i] : 0.0f;
}

// ------------------------------ prologue -----------------------------------

// Per token: y_t, target, ref-grid distances, LN(x_raw) -> f16 A (K 42->64)
__global__ void prep_kernel(const float* __restrict__ pos,
                            const float* __restrict__ yseq,
                            const float* __restrict__ noise,
                            const float* __restrict__ t,
                            const float* __restrict__ g,
                            const float* __restrict__ bb,
                            half_t* __restrict__ Aout,
                            float* __restrict__ target) {
  int m = blockIdx.x * blockDim.x + threadIdx.x;
  if (m >= kBN) return;
  int b = m >> 12;
  float tb = t[b];
  float xr[kINRAW];
  float px = pos[m * 2 + 0], py = pos[m * 2 + 1];
  xr[0] = px; xr[1] = py;
#pragma unroll
  for (int o = 0; o < kOUT; ++o) {
    float y = yseq[m * kOUT + o], nz = noise[m * kOUT + o];
    xr[2 + o] = tb * y + (1.0f - tb) * nz;
    target[m * kOUT + o] = y - nz;
  }
  const float gc[4] = {0.0f, 1.0f / 3.0f, 2.0f / 3.0f, 1.0f};
#pragma unroll
  for (int i = 0; i < 4; ++i)
#pragma unroll
    for (int j = 0; j < 4; ++j) {
      float dx = px - gc[i], dy = py - gc[j];
      xr[26 + i * 4 + j] = sqrtf(dx * dx + dy * dy) + 1e-8f;
    }
  float mean = 0.f;
#pragma unroll
  for (int i = 0; i < kINRAW; ++i) mean += xr[i];
  mean *= (1.0f / kINRAW);
  float var = 0.f;
#pragma unroll
  for (int i = 0; i < kINRAW; ++i) { float d = xr[i] - mean; var += d * d; }
  var *= (1.0f / kINRAW);
  float inv = rsqrtf(var + 1e-5f);
#pragma unroll
  for (int i = 0; i < kINRAW; ++i)
    Aout[(size_t)m * 64 + i] = (half_t)(((xr[i] - mean) * inv) * g[i] + bb[i]);
#pragma unroll
  for (int i = kINRAW; i < 64; ++i) Aout[(size_t)m * 64 + i] = (half_t)0.0f;
}

// fx = LN(h; pre_ln2) + placeholder + x_seq @ temp_w + temp_b   (block/token)
__global__ __launch_bounds__(256) void fx_kernel(
    const float* __restrict__ h, const float* __restrict__ g,
    const float* __restrict__ bb, const float* __restrict__ ph,
    const float* __restrict__ xseq, const float* __restrict__ tw,
    const float* __restrict__ tbias, float* __restrict__ x) {
  __shared__ float red[256];
  __shared__ float xs[kTIN];
  int m = blockIdx.x, tid = threadIdx.x;
  float v0 = h[(size_t)m * kD + tid], v1 = h[(size_t)m * kD + tid + 256];
  float mean = block_reduce_sum(v0 + v1, red, 256) * (1.0f / kD);
  float d0 = v0 - mean, d1 = v1 - mean;
  float var = block_reduce_sum(d0 * d0 + d1 * d1, red, 256) * (1.0f / kD);
  float inv = rsqrtf(var + 1e-5f);
  if (tid < kTIN) xs[tid] = xseq[(size_t)m * kTIN + tid];
  __syncthreads();
  for (int rep = 0; rep < 2; ++rep) {
    int d = tid + rep * 256;
    float val = (rep ? d1 : d0) * inv * g[d] + bb[d];
    float a = ph[d] + tbias[d];
#pragma unroll
    for (int k = 0; k < kTIN; ++k) a += xs[k] * tw[k * kD + d];
    x[(size_t)m * kD + d] = val + a;
  }
}

// LN (+optional affine) + adaLN modulation + f16 pack.  One block per token.
__global__ __launch_bounds__(256) void ln_mod_pack(
    const float* __restrict__ x, const float* __restrict__ g,
    const float* __restrict__ bb, const float* __restrict__ ada, int ada_ld,
    int sh_ofs, int sc_ofs, half_t* __restrict__ out, float eps) {
  __shared__ float red[256];
  int m = blockIdx.x, tid = threadIdx.x, b = m >> 12;
  float v0 = x[(size_t)m * kD + tid], v1 = x[(size_t)m * kD + tid + 256];
  float mean = block_reduce_sum(v0 + v1, red, 256) * (1.0f / kD);
  float d0 = v0 - mean, d1 = v1 - mean;
  float var = block_reduce_sum(d0 * d0 + d1 * d1, red, 256) * (1.0f / kD);
  float inv = rsqrtf(var + eps);
  const float* ab = ada + (size_t)b * ada_ld;
  for (int rep = 0; rep < 2; ++rep) {
    int d = tid + rep * 256;
    float nv = (rep ? d1 : d0) * inv;
    if (g) nv = nv * g[d] + bb[d];
    float sh = ab[sh_ofs + d], sc = ab[sc_ofs + d];
    out[(size_t)m * kD + d] = (half_t)(nv * (1.0f + sc) + sh);
  }
}

// s = LN(sf @ sen_w + sen_b), one block per sampled row (B*NS rows)
__global__ __launch_bounds__(256) void s_kernel(
    const float* __restrict__ pos, const float* __restrict__ xseq,
    const int* __restrict__ idx, const float* __restrict__ w,
    const float* __restrict__ bi, const float* __restrict__ lg,
    const float* __restrict__ lb, float* __restrict__ s,
    half_t* __restrict__ s16) {
  __shared__ float sf[14];
  __shared__ float red[256];
  int row = blockIdx.x, b = row >> 7, j = row & 127, tid = threadIdx.x;
  if (tid < 14) {
    int n = idx[j];
    sf[tid] = (tid < 2) ? pos[((size_t)b * kN + n) * 2 + tid]
                        : xseq[((size_t)b * kN + n) * kTIN + (tid - 2)];
  }
  __syncthreads();
  float u0 = bi[tid], u1 = bi[tid + 256];
#pragma unroll
  for (int k = 0; k < 14; ++k) {
    float f = sf[k];
    u0 += f * w[k * kD + tid];
    u1 += f * w[k * kD + tid + 256];
  }
  float mean = block_reduce_sum(u0 + u1, red, 256) * (1.0f / kD);
  float d0 = u0 - mean, d1 = u1 - mean;
  float inv = rsqrtf(block_reduce_sum(d0 * d0 + d1 * d1, red, 256) * (1.0f / kD) + 1e-5f);
  float r0 = d0 * inv * lg[tid] + lb[tid];
  float r1 = d1 * inv * lg[tid + 256] + lb[tid + 256];
  s[(size_t)row * kD + tid] = r0;
  s[(size_t)row * kD + tid + 256] = r1;
  s16[(size_t)row * kD + tid] = (half_t)r0;
  s16[(size_t)row * kD + tid + 256] = (half_t)r1;
}

__global__ __launch_bounds__(128) void s2_kernel(
    const float* __restrict__ pos, const float* __restrict__ xseq,
    const int* __restrict__ idx, const float* __restrict__ w,
    const float* __restrict__ bi, const float* __restrict__ lg,
    const float* __restrict__ lb, float* __restrict__ s2) {
  __shared__ float sf[14];
  __shared__ float red[128];
  int row = blockIdx.x, b = row >> 7, j = row & 127, tid = threadIdx.x;
  if (tid < 14) {
    int n = idx[j];
    sf[tid] = (tid < 2) ? pos[((size_t)b * kN + n) * 2 + tid]
                        : xseq[((size_t)b * kN + n) * kTIN + (tid - 2)];
  }
  __syncthreads();
  float u = bi[tid];
#pragma unroll
  for (int k = 0; k < 14; ++k) u += sf[k] * w[k * kC + tid];
  float mean = block_reduce_sum(u, red, 128) * (1.0f / kC);
  float d = u - mean;
  float inv = rsqrtf(block_reduce_sum(d * d, red, 128) * (1.0f / kC) + 1e-5f);
  s2[(size_t)row * kC + tid] = d * inv * lg[tid] + lb[tid];
}

__global__ void s2_mean(const float* __restrict__ s2, float* __restrict__ s2m) {
  int id = blockIdx.x * blockDim.x + threadIdx.x;  // b*128 + c
  if (id >= kB * kC) return;
  int b = id >> 7, c = id & 127;
  float a = 0.f;
  for (int j = 0; j < kNS; ++j) a += s2[((size_t)b * kNS + j) * kC + c];
  s2m[id] = a * (1.0f / kNS);
}

__global__ __launch_bounds__(128) void temb_kernel(
    const float* __restrict__ t, const float* __restrict__ w1,
    const float* __restrict__ b1, const float* __restrict__ w2,
    const float* __restrict__ b2, const float* __restrict__ s2m,
    float* __restrict__ mu) {
  __shared__ float emb[kC];
  __shared__ float h1[kC];
  int b = blockIdx.x, i = threadIdx.x;
  float tb = t[b];
  float fr = expf(-logf(10000.0f) * (float)(i & 63) * (1.0f / 64.0f));
  float ang = tb * fr;
  emb[i] = (i < 64) ? cosf(ang) : sinf(ang);
  __syncthreads();
  float a = b1[i];
  for (int k = 0; k < kC; ++k) a += emb[k] * w1[k * kC + i];
  h1[i] = silu_f(a);
  __syncthreads();
  float te = b2[i] + s2m[b * kC + i];
  for (int k = 0; k < kC; ++k) te += h1[k] * w2[k * kC + i];
  mu[b * kC + i] = silu_f(te);
}

// out[b,j] = mu[b,:] @ W[128,Nc] + bi[j]
__global__ __launch_bounds__(256) void ada_kernel(
    const float* __restrict__ mu, const float* __restrict__ W,
    const float* __restrict__ bi, float* __restrict__ out, int Nc) {
  __shared__ float ms[kC];
  int b = blockIdx.x, tid = threadIdx.x;
  if (tid < kC) ms[tid] = mu[b * kC + tid];
  __syncthreads();
  int j = blockIdx.y * 256 + tid;
  if (j >= Nc) return;
  float a = bi[j];
  for (int k = 0; k < kC; ++k) a += ms[k] * W[(size_t)k * Nc + j];
  out[(size_t)b * Nc + j] = a;
}

// ------------------------------ attention ----------------------------------

// Cross attention: per (b, head, query). 128 keys, head-dim 8, online softmax.
__global__ __launch_bounds__(256) void cross_attn(
    const float* __restrict__ q, const float* __restrict__ kv,
    float* __restrict__ o) {
  __shared__ float kvs[kNS * 64];  // 32 KB
  int b = blockIdx.x, tid = threadIdx.x;
  for (int i = tid; i < kNS * 64; i += 256) kvs[i] = kv[(size_t)b * kNS * 64 + i];
  __syncthreads();
  int n = blockIdx.y * 64 + (tid >> 2);
  int h = tid & 3;
  float q8[kDH];
#pragma unroll
  for (int c = 0; c < kDH; ++c) q8[c] = q[((size_t)b * kN + n) * 64 + h * kDH + c];
  const float scale = 0.35355339059327373f;  // 1/sqrt(8)
  float mx = -1e30f, sum = 0.f, acc[kDH] = {};
  for (int j = 0; j < kNS; ++j) {
    const float* kk = &kvs[j * 64 + h * kDH];
    const float* vv = &kvs[j * 64 + 32 + h * kDH];
    float sc = 0.f;
#pragma unroll
    for (int c = 0; c < kDH; ++c) sc += q8[c] * kk[c];
    sc *= scale;
    float nm = fmaxf(mx, sc);
    float corr = expf(mx - nm), e = expf(sc - nm);
    sum = sum * corr + e;
#pragma unroll
    for (int c = 0; c < kDH; ++c) acc[c] = acc[c] * corr + e * vv[c];
    mx = nm;
  }
  float invs = 1.0f / sum;
#pragma unroll
  for (int c = 0; c < kDH; ++c)
    o[((size_t)b * kN + n) * 32 + h * kDH + c] = acc[c] * invs;
}

// softmax over G=32 slots for every (b,h,n)
__global__ __launch_bounds__(256) void pa_softmax(
    const float* __restrict__ xm, const float* __restrict__ wsl,
    const float* __restrict__ bsl, const float* __restrict__ temp,
    float* __restrict__ sw) {
  __shared__ float w[kDH * kG];
  __shared__ float bs[kG];
  __shared__ float tp[kHP];
  int tid = threadIdx.x;
  w[tid] = wsl[tid];  // 256 == 8*32
  if (tid < kG) bs[tid] = bsl[tid];
  if (tid < kHP) tp[tid] = temp[tid];
  __syncthreads();
  int gid = blockIdx.x * 256 + tid;  // (b*8+h)*4096 + n
  int n = gid & (kN - 1);
  int bh = gid >> 12;
  int b = bh >> 3, h = bh & 7;
  float xv[kDH];
#pragma unroll
  for (int c = 0; c < kDH; ++c) xv[c] = xm[((size_t)b * kN + n) * 64 + h * kDH + c];
  float it = 1.0f / tp[h];
  float lg[kG];
  float mx = -1e30f;
#pragma unroll
  for (int g = 0; g < kG; ++g) {
    float a = bs[g];
#pragma unroll
    for (int c = 0; c < kDH; ++c) a += xv[c] * w[c * kG + g];
    a *= it;
    lg[g] = a;
    mx = fmaxf(mx, a);
  }
  float s = 0.f;
#pragma unroll
  for (int g = 0; g < kG; ++g) { lg[g] = expf(lg[g] - mx); s += lg[g]; }
  float inv = 1.0f / s;
#pragma unroll
  for (int g = 0; g < kG; ++g) sw[(size_t)gid * kG + g] = lg[g] * inv;
}

// tok[b,h,g,:] = (sum_n fxm * sw) / (sum_n sw + 1e-5)   (one block per b,h,g)
__global__ __launch_bounds__(256) void tok_kernel(
    const float* __restrict__ fxm, const float* __restrict__ sw,
    float* __restrict__ tok) {
  __shared__ float red[256];
  int blk = blockIdx.x;
  int g = blk & (kG - 1);
  int bh = blk >> 5;
  int b = bh >> 3, h = bh & 7;
  int tid = threadIdx.x;
  float acc[kDH] = {};
  float sn = 0.f;
  for (int n = tid; n < kN; n += 256) {
    float wv = sw[((size_t)bh * kN + n) * kG + g];
    sn += wv;
    const float* f = &fxm[((size_t)b * kN + n) * 64 + h * kDH];
#pragma unroll
    for (int c = 0; c < kDH; ++c) acc[c] += f[c] * wv;
  }
  float snt = block_reduce_sum(sn, red, 256);
  float inv = 1.0f / (snt + 1e-5f);
#pragma unroll
  for (int c = 0; c < kDH; ++c) {
    float tot = block_reduce_sum(acc[c], red, 256);
    if (tid == 0) tok[(size_t)blk * kDH + c] = tot * inv;
  }
}

// tiny self-attention over the 32 cluster tokens (one wave per b,h)
__global__ __launch_bounds__(32) void tok_attn(
    const float* __restrict__ tok, const float* __restrict__ wq,
    const float* __restrict__ wk, const float* __restrict__ wv,
    float* __restrict__ os) {
  __shared__ float ks[kG][kDH];
  __shared__ float vs[kG][kDH];
  int bh = blockIdx.x, g = threadIdx.x;
  float t8[kDH];
#pragma unroll
  for (int c = 0; c < kDH; ++c) t8[c] = tok[((size_t)bh * kG + g) * kDH + c];
  float q8[kDH];
#pragma unroll
  for (int c = 0; c < kDH; ++c) {
    float aq = 0.f, ak = 0.f, av = 0.f;
#pragma unroll
    for (int d = 0; d < kDH; ++d) {
      aq += t8[d] * wq[d * kDH + c];
      ak += t8[d] * wk[d * kDH + c];
      av += t8[d] * wv[d * kDH + c];
    }
    q8[c] = aq;
    ks[g][c] = ak;
    vs[g][c] = av;
  }
  __syncthreads();
  const float scale = 0.35355339059327373f;
  float sc[kG];
  float mx = -1e30f;
#pragma unroll
  for (int g2 = 0; g2 < kG; ++g2) {
    float a = 0.f;
#pragma unroll
    for (int c = 0; c < kDH; ++c) a += q8[c] * ks[g2][c];
    a *= scale;
    sc[g2] = a;
    mx = fmaxf(mx, a);
  }
  float s = 0.f;
#pragma unroll
  for (int g2 = 0; g2 < kG; ++g2) { sc[g2] = expf(sc[g2] - mx); s += sc[g2]; }
  float inv = 1.0f / s;
  float acc[kDH] = {};
#pragma unroll
  for (int g2 = 0; g2 < kG; ++g2) {
    float p = sc[g2] * inv;
#pragma unroll
    for (int c = 0; c < kDH; ++c) acc[c] += p * vs[g2][c];
  }
#pragma unroll
  for (int c = 0; c < kDH; ++c) os[((size_t)bh * kG + g) * kDH + c] = acc[c];
}

// ox[b,n,h,:] = sum_g os[b,h,g,:] * sw[b,h,n,g]
__global__ __launch_bounds__(256) void pa_ox(
    const float* __restrict__ os, const float* __restrict__ sw,
    float* __restrict__ ox) {
  __shared__ float oss[kG * kDH];
  int bh = blockIdx.x;
  int tid = threadIdx.x;
  oss[tid] = os[(size_t)bh * kG * kDH + tid];
  __syncthreads();
  int n = blockIdx.y * 256 + tid;
  int b = bh >> 3, h = bh & 7;
  float acc[kDH] = {};
#pragma unroll
  for (int g = 0; g < kG; ++g) {
    float wv = sw[((size_t)bh * kN + n) * kG + g];
#pragma unroll
    for (int c = 0; c < kDH; ++c) acc[c] += oss[g * kDH + c] * wv;
  }
#pragma unroll
  for (int c = 0; c < kDH; ++c)
    ox[((size_t)b * kN + n) * 64 + h * kDH + c] = acc[c];
}

// ------------------------------- loss --------------------------------------

__global__ __launch_bounds__(256) void loss_partial(
    const float* __restrict__ pred, const float* __restrict__ target,
    float* __restrict__ partial) {
  __shared__ float red[256];
  int m = blockIdx.x * 256 + threadIdx.x;
  float a = 0.f;
#pragma unroll
  for (int o = 0; o < kOUT; ++o) {
    float d = pred[(size_t)m * 64 + o] - target[(size_t)m * kOUT + o];
    a += d * d;
  }
  float s = block_reduce_sum(a, red, 256);
  if (threadIdx.x == 0) partial[blockIdx.x] = s;
}

__global__ __launch_bounds__(64) void loss_final(const float* __restrict__ partial,
                                                 float* __restrict__ out) {
  __shared__ float red[64];
  float v = partial[threadIdx.x];
  float s = block_reduce_sum(v, red, 64);
  if (threadIdx.x == 0) out[0] = s * (1.0f / ((float)kBN * (float)kOUT));
}

// ----------------------------- input layout --------------------------------

enum {
  IN_POS = 0, IN_XSEQ, IN_YSEQ, IN_T, IN_NOISE, IN_IDX,
  P_PRE_LN1_G, P_PRE_LN1_B, P_PRE_W, P_PRE_B, P_PRE_LN2_G, P_PRE_LN2_B,
  P_PLACEHOLDER, P_TEMP_W, P_TEMP_B, P_SEN_W, P_SEN_B, P_SEN_LN_G, P_SEN_LN_B,
  P_SEN2_W, P_SEN2_B, P_SEN2_LN_G, P_SEN2_LN_B,
  P_TEMB_W1, P_TEMB_B1, P_TEMB_W2, P_TEMB_B2,
  P_FIN_ADA_W, P_FIN_ADA_B, P_FIN_W1, P_FIN_B1, P_FIN_W2, P_FIN_B2,
  L_ADA1_W, L_ADA1_B, L_ADA2_W, L_ADA2_B,
  L_N1_G, L_N1_B, L_N2_G, L_N2_B, L_N3_G, L_N3_B, L_N4_G, L_N4_B,
  L_CA_WQ, L_CA_WKV, L_CA_WO, L_CA_BO,
  L_FF1_W1, L_FF1_B1, L_FF1_W2, L_FF1_B2,
  L_FF2_W1, L_FF2_B1, L_FF2_W2, L_FF2_B2,
  L_PA_WFX, L_PA_BFX, L_PA_WX, L_PA_BX, L_PA_WSL, L_PA_BSL, L_PA_TEMP,
  L_PA_WQ, L_PA_WK, L_PA_WV, L_PA_WO, L_PA_BO
};

// -------------------------------- driver -----------------------------------

extern "C" void kernel_launch(void* const* d_in, const int* in_sizes, int n_in,
                              void* d_out, int out_size, void* d_ws, size_t ws_size,
                              hipStream_t stream) {
  (void)in_sizes; (void)n_in; (void)out_size; (void)ws_size;

  const float* pos   = (const float*)d_in[IN_POS];
  const float* xseq  = (const float*)d_in[IN_XSEQ];
  const float* yseq  = (const float*)d_in[IN_YSEQ];
  const float* tvec  = (const float*)d_in[IN_T];
  const float* noise = (const float*)d_in[IN_NOISE];
  const int*   idx   = (const int*)d_in[IN_IDX];
  auto P = [&](int i) { return (const float*)d_in[i]; };
  float* out = (float*)d_out;

  char* ws = (char*)d_ws;
  size_t off = 0;
  auto allocf = [&](size_t n) { float* p = (float*)(ws + off); off = (off + n * 4 + 255) & ~(size_t)255; return p; };
  auto alloch = [&](size_t n) { half_t* p = (half_t*)(ws + off); off = (off + n * 2 + 255) & ~(size_t)255; return p; };

  half_t* Axr    = alloch((size_t)kBN * 64);
  float*  target = allocf((size_t)kBN * kOUT);
  float*  x      = allocf((size_t)kBN * kD);
  float*  tmp    = allocf((size_t)kBN * kD);
  half_t* hA     = alloch((size_t)kBN * kD);
  half_t* hB     = alloch((size_t)kBN * kD);
  float*  qb     = allocf((size_t)kBN * 64);
  float*  kvb    = allocf((size_t)kB * kNS * 64);
  float*  ob     = allocf((size_t)kBN * 32);
  half_t* o16    = alloch((size_t)kBN * 32);
  float*  fxmb   = allocf((size_t)kBN * 64);
  float*  xmb    = allocf((size_t)kBN * 64);
  float*  swb    = allocf((size_t)kB * kHP * kN * kG);
  float*  tokb   = allocf((size_t)kB * kHP * kG * kDH);
  float*  osb    = allocf((size_t)kB * kHP * kG * kDH);
  float*  oxb    = allocf((size_t)kBN * 64);
  half_t* ox16   = alloch((size_t)kBN * 64);
  float*  sbuf   = allocf((size_t)kB * kNS * kD);
  half_t* s16    = alloch((size_t)kB * kNS * kD);
  float*  s2buf  = allocf((size_t)kB * kNS * kC);
  float*  s2m    = allocf((size_t)kB * kC);
  float*  mub    = allocf((size_t)kB * kC);
  float*  ada1b  = allocf((size_t)kB * 6 * kD);
  float*  ada2b  = allocf((size_t)kB * 6 * kD);
  float*  finada = allocf((size_t)kB * 2 * kD);
  float*  bias64 = allocf(64);
  float*  lpart  = allocf(64);
  half_t* wpre   = alloch((size_t)512 * 64);
  half_t* wfin1  = alloch((size_t)512 * 512);
  half_t* wfin2  = alloch((size_t)64 * 512);
  half_t* wq16   = alloch((size_t)64 * 512);
  half_t* wkv16  = alloch((size_t)64 * 512);
  half_t* wo16   = alloch((size_t)512 * 32);
  half_t* wf11   = alloch((size_t)512 * 512);
  half_t* wf12   = alloch((size_t)512 * 512);
  half_t* wf21   = alloch((size_t)512 * 512);
  half_t* wf22   = alloch((size_t)512 * 512);
  half_t* wfx16  = alloch((size_t)64 * 512);
  half_t* wx16   = alloch((size_t)64 * 512);
  half_t* wpo16  = alloch((size_t)512 * 64);

  auto packW = [&](const float* W, half_t* Wt, int K, int Ns, int Kp, int Np) {
    int tot = Np * Kp;
    pack_wt<<<(tot + 255) / 256, 256, 0, stream>>>(W, Wt, K, Ns, Kp, Np);
  };
  auto gemmF = [&](const half_t* A, const half_t* Bt, const float* bias,
                   float* Cf, int M, int N, int K) {
    dim3 g(M / 64, N / 64);
    gemm_wmma<0><<<g, 128, 0, stream>>>(A, Bt, bias, Cf, nullptr, nullptr,
                                        nullptr, 0, 0, M, N, K);
  };
  auto gemmG = [&](const half_t* A, const half_t* Bt, const float* bias,
                   half_t* Ch, int M, int N, int K) {
    dim3 g(M / 64, N / 64);
    gemm_wmma<1><<<g, 128, 0, stream>>>(A, Bt, bias, nullptr, Ch, nullptr,
                                        nullptr, 0, 0, M, N, K);
  };
  auto gemmR = [&](const half_t* A, const half_t* Bt, const float* bias,
                   float* resid, const float* gate, int gofs, int M, int N, int K) {
    dim3 g(M / 64, N / 64);
    gemm_wmma<2><<<g, 128, 0, stream>>>(A, Bt, bias, nullptr, nullptr, resid,
                                        gate, 6 * kD, gofs, M, N, K);
  };

  // -------- prologue --------
  prep_kernel<<<kBN / 256, 256, 0, stream>>>(pos, yseq, noise, tvec,
                                             P(P_PRE_LN1_G), P(P_PRE_LN1_B),
                                             Axr, target);
  packW(P(P_PRE_W), wpre, kINRAW, kD, 64, kD);
  packW(P(P_FIN_W1), wfin1, kD, kD, kD, kD);
  packW(P(P_FIN_W2), wfin2, kD, kOUT, kD, 64);
  pad_bias<<<1, 64, 0, stream>>>(P(P_FIN_B2), bias64, kOUT, 64);

  gemmF(Axr, wpre, P(P_PRE_B), tmp, kBN, kD, 64);
  fx_kernel<<<kBN, 256, 0, stream>>>(tmp, P(P_PRE_LN2_G), P(P_PRE_LN2_B),
                                     P(P_PLACEHOLDER), xseq, P(P_TEMP_W),
                                     P(P_TEMP_B), x);
  s_kernel<<<kB * kNS, 256, 0, stream>>>(pos, xseq, idx, P(P_SEN_W), P(P_SEN_B),
                                         P(P_SEN_LN_G), P(P_SEN_LN_B), sbuf, s16);
  s2_kernel<<<kB * kNS, 128, 0, stream>>>(pos, xseq, idx, P(P_SEN2_W),
                                          P(P_SEN2_B), P(P_SEN2_LN_G),
                                          P(P_SEN2_LN_B), s2buf);
  s2_mean<<<(kB * kC + 255) / 256, 256, 0, stream>>>(s2buf, s2m);
  temb_kernel<<<kB, 128, 0, stream>>>(tvec, P(P_TEMB_W1), P(P_TEMB_B1),
                                      P(P_TEMB_W2), P(P_TEMB_B2), s2m, mub);

  // -------- layers --------
  for (int l = 0; l < kL; ++l) {
    auto LPW = [&](int which, size_t stride) { return P(which) + (size_t)l * stride; };
    const float* ada1_w = LPW(L_ADA1_W, (size_t)kC * 6 * kD);
    const float* ada1_b = LPW(L_ADA1_B, 6 * kD);
    const float* ada2_w = LPW(L_ADA2_W, (size_t)kC * 6 * kD);
    const float* ada2_b = LPW(L_ADA2_B, 6 * kD);

    ada_kernel<<<dim3(kB, (6 * kD) / 256), 256, 0, stream>>>(mub, ada1_w, ada1_b, ada1b, 6 * kD);
    ada_kernel<<<dim3(kB, (6 * kD) / 256), 256, 0, stream>>>(mub, ada2_w, ada2_b, ada2b, 6 * kD);

    packW(LPW(L_CA_WQ,  (size_t)kD * 32), wq16, kD, 32, kD, 64);
    packW(LPW(L_CA_WKV, (size_t)kD * 64), wkv16, kD, 64, kD, 64);
    packW(LPW(L_CA_WO,  (size_t)32 * kD), wo16, 32, kD, 32, kD);
    packW(LPW(L_FF1_W1, (size_t)kD * kD), wf11, kD, kD, kD, kD);
    packW(LPW(L_FF1_W2, (size_t)kD * kD), wf12, kD, kD, kD, kD);
    packW(LPW(L_FF2_W1, (size_t)kD * kD), wf21, kD, kD, kD, kD);
    packW(LPW(L_FF2_W2, (size_t)kD * kD), wf22, kD, kD, kD, kD);
    packW(LPW(L_PA_WFX, (size_t)kD * 64), wfx16, kD, 64, kD, 64);
    packW(LPW(L_PA_WX,  (size_t)kD * 64), wx16, kD, 64, kD, 64);
    packW(LPW(L_PA_WO,  (size_t)64 * kD), wpo16, 64, kD, 64, kD);

    // cross-attention branch
    ln_mod_pack<<<kBN, 256, 0, stream>>>(x, LPW(L_N1_G, kD), LPW(L_N1_B, kD),
                                         ada1b, 6 * kD, 0, kD, hA, 1e-5f);
    gemmF(hA, wq16, nullptr, qb, kBN, 64, kD);
    gemmF(s16, wkv16, nullptr, kvb, kB * kNS, 64, kD);
    cross_attn<<<dim3(kB, kN / 64), 256, 0, stream>>>(qb, kvb, ob);
    pack_a<<<(kBN * 32 + 255) / 256, 256, 0, stream>>>(ob, o16, kBN, 32, 32);
    gemmR(o16, wo16, LPW(L_CA_BO, kD), x, ada1b, 2 * kD, kBN, kD, 32);

    // FFN 1
    ln_mod_pack<<<kBN, 256, 0, stream>>>(x, LPW(L_N2_G, kD), LPW(L_N2_B, kD),
                                         ada1b, 6 * kD, 3 * kD, 4 * kD, hA, 1e-5f);
    gemmG(hA, wf11, LPW(L_FF1_B1, kD), hB, kBN, kD, kD);
    gemmR(hB, wf12, LPW(L_FF1_B2, kD), x, ada1b, 5 * kD, kBN, kD, kD);

    // cluster ("pa") branch
    ln_mod_pack<<<kBN, 256, 0, stream>>>(x, LPW(L_N3_G, kD), LPW(L_N3_B, kD),
                                         ada2b, 6 * kD, 0, kD, hA, 1e-5f);
    gemmF(hA, wfx16, LPW(L_PA_BFX, 64), fxmb, kBN, 64, kD);
    gemmF(hA, wx16, LPW(L_PA_BX, 64), xmb, kBN, 64, kD);
    pa_softmax<<<(kB * kHP * kN) / 256, 256, 0, stream>>>(
        xmb, LPW(L_PA_WSL, kDH * kG), LPW(L_PA_BSL, kG), LPW(L_PA_TEMP, kHP), swb);
    tok_kernel<<<kB * kHP * kG, 256, 0, stream>>>(fxmb, swb, tokb);
    tok_attn<<<kB * kHP, 32, 0, stream>>>(tokb, LPW(L_PA_WQ, kDH * kDH),
                                          LPW(L_PA_WK, kDH * kDH),
                                          LPW(L_PA_WV, kDH * kDH), osb);
    pa_ox<<<dim3(kB * kHP, kN / 256), 256, 0, stream>>>(osb, swb, oxb);
    pack_a<<<(kBN * 64 + 255) / 256, 256, 0, stream>>>(oxb, ox16, kBN, 64, 64);
    gemmR(ox16, wpo16, LPW(L_PA_BO, kD), x, ada2b, 2 * kD, kBN, kD, 64);

    // FFN 2
    ln_mod_pack<<<kBN, 256, 0, stream>>>(x, LPW(L_N4_G, kD), LPW(L_N4_B, kD),
                                         ada2b, 6 * kD, 3 * kD, 4 * kD, hA, 1e-5f);
    gemmG(hA, wf21, LPW(L_FF2_B1, kD), hB, kBN, kD, kD);
    gemmR(hB, wf22, LPW(L_FF2_B2, kD), x, ada2b, 5 * kD, kBN, kD, kD);
  }

  // -------- head + loss --------
  ada_kernel<<<dim3(kB, (2 * kD) / 256), 256, 0, stream>>>(mub, P(P_FIN_ADA_W),
                                                           P(P_FIN_ADA_B), finada, 2 * kD);
  ln_mod_pack<<<kBN, 256, 0, stream>>>(x, nullptr, nullptr, finada, 2 * kD,
                                       0, kD, hA, 1e-6f);
  gemmG(hA, wfin1, P(P_FIN_B1), hB, kBN, kD, kD);
  gemmF(hB, wfin2, bias64, tmp, kBN, 64, kD);  // pred, ld = 64
  loss_partial<<<kBN / 256, 256, 0, stream>>>(tmp, target, lpart);
  loss_final<<<1, 64, 0, stream>>>(lpart, out);
}
